// Conv2dLocal_77532749627514
// MI455X (gfx1250) — matrix operand order repaired and checked by reading, not compiled, not run on hardware
//
#include <hip/hip_runtime.h>

// ---------------------------------------------------------------------------
// Fixed-point (f=8) 3x3 conv, NHWC 32x224x224x16 -> 32x222x222x16.
// Exact-integer semantics via V_WMMA_F32_16X16X32_F16:
//   ix=round(x*2^f), iw=round(w*2^f) exact in f16 (|ix|<2048, |iw|<128),
//   acc = sum(ix*iw) < 2^24 exact in f32 accumulator,
//   out = floor(acc*2^-f)*2^-f (exact power-of-two scaling).
// Each block: 6 output rows x 256-wide row tile; 8 input rows staged once.
// A/B LDS accesses are indexed off the __shared__ arrays with integer offsets
// only (no pointer PHIs), so they stay addrspace(3) -> ds_load_b128.
// ---------------------------------------------------------------------------

typedef _Float16 v16h __attribute__((ext_vector_type(16)));
typedef _Float16 v8h  __attribute__((ext_vector_type(8)));
typedef float    v8f  __attribute__((ext_vector_type(8)));

#define H_IN   224
#define W_IN   224
#define C_IN   16
#define C_OUT  16
#define H_OUT  222
#define W_OUT  222
#define KH     3
#define KW     3
#define TILE_W 256                // one tile spans the whole output row
#define XROW   (TILE_W + KW - 1)  // 258 staged input pixels per row
#define RROW   (XROW * C_IN)      // halfs per staged input row (4128)
#define KREAL  (KH * KW * C_IN)   // 144
#define KPAD   160                // 5 chunks of K=32
#define NTHREADS 128
#define MSUB   4                  // M-subtiles (16 pixels each) per wave
#define HC     6                  // output rows per block (222 = 37*6 exact)
#define NROWS  (HC + KH - 1)      // 8 staged input rows (h0+7 <= 223 always)

// ---- prep: quantize + transpose + pad weights into ws as [n][KPAD] f16 ----
__global__ void quant_weights_kernel(const float* __restrict__ w,
                                     const int* __restrict__ fp,
                                     _Float16* __restrict__ bq) {
    const int i = blockIdx.x * blockDim.x + threadIdx.x;
    if (i >= C_OUT * KPAD) return;
    const int n = i / KPAD;
    const int k = i - n * KPAD;
    const float S = (float)(1 << fp[0]);
    // HWIO flat: w[k*16 + n], k = (kh*3+kw)*16 + c
    bq[i] = (k < KREAL) ? (_Float16)rintf(w[k * C_OUT + n] * S) : (_Float16)0.0f;
}

__global__ __launch_bounds__(NTHREADS) void conv_fp_wmma(
    const float*    __restrict__ x,   // [32,224,224,16] NHWC
    const float*    __restrict__ w,   // [3,3,16,16] HWIO (fallback path)
    const int*      __restrict__ fp,  // scalar fixed_point
    const _Float16* __restrict__ bq,  // prequantized weights [n][KPAD] (ws path)
    int use_ws,
    float*          __restrict__ out) // [32,222,222,16]
{
    __shared__ __align__(16) _Float16 lx[NROWS * RROW];   // 66048 B
    __shared__ __align__(16) _Float16 lb[C_OUT * KPAD];   //  5120 B

    const int tid = threadIdx.x;
    const int f   = fp[0];
    const float S    = (float)(1 << f);
    const float invS = __int_as_float((127 - f) << 23);   // exact 2^-f, no v_div

    const int bid   = blockIdx.x;
    const int hc    = bid % (H_OUT / HC);
    const int n_img = bid / (H_OUT / HC);
    const int h0    = hc * HC;

    // ---- stage NROWS quantized input rows into LDS (no row guards needed) ----
    for (int r = 0; r < NROWS; ++r) {
        const float* row = x + (((size_t)n_img * H_IN + (h0 + r)) * W_IN) * C_IN;
        const int dbase = r * RROW;
        for (int i = tid * 4; i < RROW; i += NTHREADS * 4) {
            float4 v = make_float4(0.f, 0.f, 0.f, 0.f);
            if ((i >> 4) < W_IN) v = *(const float4*)(row + i);
            lx[dbase + i + 0] = (_Float16)rintf(v.x * S);  // v_rndne: RNE = jnp.round
            lx[dbase + i + 1] = (_Float16)rintf(v.y * S);
            lx[dbase + i + 2] = (_Float16)rintf(v.z * S);
            lx[dbase + i + 3] = (_Float16)rintf(v.w * S);
        }
    }

    // ---- stage weights: linear 5KB copy from ws (fast path) or quantize here ----
    if (use_ws) {
        const uint4* src = (const uint4*)bq;
        uint4*       dst = (uint4*)lb;
        for (int i = tid; i < (C_OUT * KPAD) / 8; i += NTHREADS)  // 320 x 16B
            dst[i] = src[i];
    } else {
        for (int i = tid; i < C_OUT * (KPAD - KREAL); i += NTHREADS) {
            const int n = i / (KPAD - KREAL);
            const int k = KREAL + i % (KPAD - KREAL);
            lb[n * KPAD + k] = (_Float16)0.0f;
        }
        for (int i = tid; i < KREAL * C_OUT; i += NTHREADS) {
            const int n = i & (C_OUT - 1);
            const int k = i >> 4;
            lb[n * KPAD + k] = (_Float16)rintf(w[i] * S);
        }
    }
    __syncthreads();

    // ---- each wave: four 16(M)x16(N) D tiles sharing each B fragment ----
    const int wid   = tid >> 5;
    const int lane  = tid & 31;
    const int nn    = lane & 15;      // A: row index; B/C/D: column N
    const int g     = lane >> 4;      // lane half
    const int mbase = wid * (16 * MSUB);

    // A-fragment addressing: for chunk kc, lo run starts at K=kc*32+g*8,
    // hi run at K=kc*32+16+g*8; no run crosses a 48-element kh slice, and the
    // g=1 run is always +8 halfs after the g=0 run:
    //   half-index = abase0 + hh*RROW + s*256 + CONST(kc,part)
    constexpr int ALO[5] = { 0, 32, RROW + 16, 2 * RROW,      2 * RROW + 32 };
    constexpr int AHI[5] = { 16, RROW, RROW + 32, 2 * RROW + 16, -1 /*pad*/ };

    const int abase0 = (mbase + nn) * C_IN + g * 8;   // per-lane, computed once
    const int bbase  = nn * KPAD + g * 8;

    size_t obase = (((size_t)n_img * H_OUT + h0) * W_OUT) * C_OUT + nn;
    int ab = abase0;

    for (int hh = 0; hh < HC; ++hh) {
        v8f acc[MSUB] = {};

#pragma unroll
        for (int kc = 0; kc < 5; ++kc) {
            // B fragment, 32x16: lane-half g holds K = kc*32 + g*8 + [0,8) and kc*32 + 16 + g*8 + [0,8);
            // loop-invariant across hh -> compiler hoists all 5 into registers.
            v16h b;
            {
                const v8h blo = *(const v8h*)&lb[bbase + kc * 32];
                const v8h bhi = *(const v8h*)&lb[bbase + kc * 32 + 16];
#pragma unroll
                for (int t = 0; t < 8; ++t) { b[t] = blo[t]; b[8 + t] = bhi[t]; }
            }
#pragma unroll
            for (int s = 0; s < MSUB; ++s) {
                const v8h lo = *(const v8h*)&lx[ab + s * 256 + ALO[kc]]; // ds_b128 imm
                v8h hi = {};
                if (AHI[kc] >= 0)                                        // folds per kc
                    hi = *(const v8h*)&lx[ab + s * 256 + AHI[kc]];
                v16h a;
#pragma unroll
                for (int t = 0; t < 8; ++t) { a[t] = lo[t]; a[8 + t] = hi[t]; }
                // (neg_a, A, neg_b, B, c_mod, C, reuse_a, reuse_b)
                acc[s] = __builtin_amdgcn_wmma_f32_16x16x32_f16(
                    false, a, false, b, (short)0, acc[s], false, false);
            }
        }

        // ---- D store, C/D layout: VGPR r -> (M = g*8 + r, N = nn) ----
#pragma unroll
        for (int s = 0; s < MSUB; ++s) {
            const int wb16 = mbase + s * 16;            // wave-uniform subtile base
            if (wb16 + 15 < W_OUT) {                    // full subtile: no exec masks
#pragma unroll
                for (int r = 0; r < 8; ++r)
                    out[obase + (size_t)(wb16 + g * 8 + r) * C_OUT] =
                        floorf(acc[s][r] * invS) * invS;
            } else if (wb16 < W_OUT) {                  // straddling subtile
#pragma unroll
                for (int r = 0; r < 8; ++r) {
                    const int wout = wb16 + g * 8 + r;
                    if (wout < W_OUT)
                        out[obase + (size_t)wout * C_OUT] =
                            floorf(acc[s][r] * invS) * invS;
                }
            }                                            // fully OOB: skip
        }

        // advance one output row: A window slides by one staged input row
        ab += RROW;
        obase += (size_t)W_OUT * C_OUT;
    }
}

extern "C" void kernel_launch(void* const* d_in, const int* in_sizes, int n_in,
                              void* d_out, int out_size, void* d_ws, size_t ws_size,
                              hipStream_t stream) {
    const float* x  = (const float*)d_in[0];
    const float* w  = (const float*)d_in[1];
    const int*   fp = (const int*)d_in[2];
    float*       out = (float*)d_out;

    _Float16* bq = (_Float16*)d_ws;
    const int use_ws = (ws_size >= (size_t)(C_OUT * KPAD) * sizeof(_Float16)) ? 1 : 0;
    if (use_ws) {
        quant_weights_kernel<<<(C_OUT * KPAD + 127) / 128, 128, 0, stream>>>(w, fp, bq);
    }

    const int n_batch = in_sizes[0] / (H_IN * W_IN * C_IN);   // 32
    const int grid = n_batch * (H_OUT / HC);                  // (n, h-chunk)
    conv_fp_wmma<<<grid, NTHREADS, 0, stream>>>(x, w, fp, bq, use_ws, out);
}
